// NTNN_81973745811971
// MI455X (gfx1250) — matrix operand order repaired and checked by reading, not compiled, not run on hardware
//
#include <hip/hip_runtime.h>
#include <hip/hip_bf16.h>
#include <math.h>

typedef _Float16 v16h __attribute__((ext_vector_type(16)));
typedef _Float16 v8h  __attribute__((ext_vector_type(8)));
typedef float    v8f  __attribute__((ext_vector_type(8)));

#define S_TILES 16
#define NPTS    32768
#define NF      128
#define HID     128
#define KFEAT   512   // 4*NF fourier features
// swizzled fragment pools in d_ws (halves)
#define W1_FRAG_HALVES (S_TILES * 8 * 16 * 512)   // s * jtile * kstep * (32 lanes * 16 halves)
#define W2_FRAG_HALVES (S_TILES * 8 * 4  * 512)
#define CHUNK_HALVES   8192                        // 16 KB staging chunk

// ---------------------------------------------------------------------------
// Kernel 1: swizzle W1 (S,512,128) and W2 (S,128,128) f32 -> f16 B-fragments.
// B-fragment layout (16-bit, 32x16): lane L -> column N = L&15,
// half i -> K = 32*t + 8*(L>>4) + i + (i&8)   (per CDNA5 ISA B-matrix striping).
// ---------------------------------------------------------------------------
__global__ void prep_weights(const float* __restrict__ W1,
                             const float* __restrict__ W2,
                             _Float16* __restrict__ ws1,
                             _Float16* __restrict__ ws2) {
    int id = blockIdx.x * blockDim.x + threadIdx.x;
    if (id < W1_FRAG_HALVES) {
        int i = id & 15;
        int L = (id >> 4) & 31;
        int t = (id >> 9) & 15;
        int j = (id >> 13) & 7;
        int s = id >> 16;
        int k = 32 * t + ((L >> 4) << 3) + i + (i & 8);      // feature index 0..511
        int h = 16 * j + (L & 15);
        ws1[id] = (_Float16)W1[(s * KFEAT + k) * HID + h];
    } else {
        int id2 = id - W1_FRAG_HALVES;
        if (id2 < W2_FRAG_HALVES) {
            int i = id2 & 15;
            int L = (id2 >> 4) & 31;
            int t = (id2 >> 9) & 3;
            int j = (id2 >> 11) & 7;
            int s = id2 >> 14;
            int k = 32 * t + ((L >> 4) << 3) + i + (i & 8);  // hidden index 0..127
            int h = 16 * j + (L & 15);
            ws2[id2] = (_Float16)W2[(s * HID + k) * HID + h];
        }
    }
}

// ---------------------------------------------------------------------------
// Async DMA of one 16 KB fragment chunk into LDS (whole 128-thread block).
// Each thread issues 8 x b128 async transfers (1024 * 16B = 16 KB).
// ---------------------------------------------------------------------------
__device__ __forceinline__ void stage_chunk16k(const _Float16* __restrict__ src,
                                               const _Float16* dstLDS, int tid) {
#pragma unroll
    for (int q = 0; q < 8; ++q) {
        int off = (q * 128 + tid) * 8;   // halves; 16-byte granules
        unsigned int       lva = (unsigned int)(unsigned long long)(size_t)(dstLDS + off);
        unsigned long long gva = (unsigned long long)(size_t)(src + off);
        asm volatile("global_load_async_to_lds_b128 %0, %1, off"
                     :: "v"(lva), "v"(gva) : "memory");
    }
}

__device__ __forceinline__ void wait_async_and_sync() {
    asm volatile("s_wait_asynccnt 0x0" ::: "memory");
    __syncthreads();
}

// ---------------------------------------------------------------------------
// Kernel 2: per (tile s, 16-point block) MLP via WMMA. One wave32 per unit,
// 4 waves (same tile) per workgroup sharing double-buffered LDS weights.
// ---------------------------------------------------------------------------
__global__ __launch_bounds__(128)
void mlp_tiles(const float* __restrict__ x,
               const float* __restrict__ Bmat,
               const _Float16* __restrict__ ws1,
               const float* __restrict__ b1,
               const _Float16* __restrict__ ws2,
               const float* __restrict__ b2,
               const float* __restrict__ W3,
               const float* __restrict__ b3,
               const float* __restrict__ mids,
               const float* __restrict__ widths,
               float* __restrict__ tau_out) {
    __shared__ _Float16 wbuf[2][CHUNK_HALVES];   // 2 x 16 KB weight staging
    __shared__ _Float16 hstage[4][16 * HID];     // 4 x 4 KB h1 transpose staging

    const int tid   = threadIdx.x;               // 0..127
    const int lane  = tid & 31;
    const int wv    = tid >> 5;                  // 0..3
    const int unit  = blockIdx.x * 4 + wv;       // 16 tiles * 2048 blocks
    const int s     = unit >> 11;                // uniform within the block
    const int pbase = (unit & 2047) * 16;
    const int m     = lane & 15;                 // A-row (point) / D-column index
    const int hi    = lane >> 4;

    const _Float16* w1src = ws1 + (size_t)s * (8 * 16 * 512);
    const _Float16* w2src = ws2 + (size_t)s * (8 * 4 * 512);

    // kick off DMA of W1 j=0 chunk; it overlaps the feature build below
    stage_chunk16k(w1src, wbuf[0], tid);

    // normalized coords for this lane's point
    const float mid = mids[s], wdt = widths[s];
    float xn[6];
#pragma unroll
    for (int c = 0; c < 6; ++c)
        xn[c] = (x[(pbase + m) * 6 + c] - mid) / wdt;

    const float TWO_PI = 6.28318530717958647692f;

    // Build 16 A-fragments of Fourier features.
    // A layout (16-bit, 16x32): lane L -> M = L&15, half i -> K = 8*(L>>4)+i+(i&8).
    v16h afrag[16];
#pragma unroll
    for (int tq = 0; tq < 4; ++tq) {
#pragma unroll
        for (int i = 0; i < 16; ++i) {
            int k = 32 * tq + (hi << 3) + i + (i & 8);   // 0..127 (NF index)
            const float* Br = &Bmat[(s * NF + k) * 3];
            float bx = Br[0], by = Br[1], bz = Br[2];
            float d0 = xn[0] * bx + xn[1] * by + xn[2] * bz;
            float d1 = xn[3] * bx + xn[4] * by + xn[5] * bz;
            float s0, c0, s1, c1;
            __sincosf(TWO_PI * d0, &s0, &c0);
            __sincosf(TWO_PI * d1, &s1, &c1);
            afrag[tq][i]      = (_Float16)s0;   // g=0: sin(2pi f0)
            afrag[4 + tq][i]  = (_Float16)c0;   // g=1: cos(2pi f0)
            afrag[8 + tq][i]  = (_Float16)s1;   // g=2: sin(2pi f1)
            afrag[12 + tq][i] = (_Float16)c1;   // g=3: cos(2pi f1)
        }
    }

    wait_async_and_sync();   // chunk 0 landed

    // ---- layer 1: feat(16x512) @ W1(512x128); 8 j-tiles, double-buffered ----
    _Float16* hbuf = hstage[wv];
    for (int j = 0; j < 8; ++j) {
        if (j < 7)
            stage_chunk16k(w1src + (j + 1) * CHUNK_HALVES, wbuf[(j + 1) & 1], tid);
        else
            stage_chunk16k(w2src, wbuf[0], tid);          // W2 first half (j2=0..3)

        const _Float16* cb = wbuf[j & 1];
        v16h wf[16];
#pragma unroll
        for (int t = 0; t < 16; ++t)
            wf[t] = *(const v16h*)&cb[t * 512 + lane * 16];
        v8f acc = {};
#pragma unroll
        for (int t = 0; t < 16; ++t)
            acc = __builtin_amdgcn_wmma_f32_16x16x32_f16(
                      false, afrag[t], false, wf[t], (short)0, acc, false, false);

        float bias = b1[s * HID + 16 * j + m];
#pragma unroll
        for (int r = 0; r < 8; ++r) {
            float hval = tanhf(acc[r] + bias);
            hbuf[(r + 8 * hi) * HID + 16 * j + m] = (_Float16)hval;  // [row][h]
        }
        wait_async_and_sync();
    }

    // reload h1 as A-fragments (LDS round-trip performs the D->A transpose;
    // this wave wrote its own staging region, LDS ops are wave-ordered)
    v16h a2[4];
#pragma unroll
    for (int t = 0; t < 4; ++t) {
        const v8h* lo  = (const v8h*)&hbuf[m * HID + 32 * t + (hi << 3)];
        const v8h* hi8 = (const v8h*)&hbuf[m * HID + 32 * t + (hi << 3) + 16];
        a2[t] = __builtin_shufflevector(*lo, *hi8,
                                        0, 1, 2, 3, 4, 5, 6, 7,
                                        8, 9, 10, 11, 12, 13, 14, 15);
    }

    // ---- layer 2: h1(16x128) @ W2(128x128), tanh, fused dot with W3 ----
    stage_chunk16k(w2src + CHUNK_HALVES, wbuf[1], tid);   // W2 second half (j2=4..7)

    float taup[8];
#pragma unroll
    for (int r = 0; r < 8; ++r) taup[r] = 0.0f;

    for (int j2 = 0; j2 < 4; ++j2) {                      // from wbuf[0]
        v8f acc = {};
#pragma unroll
        for (int t = 0; t < 4; ++t) {
            v16h bfrag = *(const v16h*)&wbuf[0][(j2 * 4 + t) * 512 + lane * 16];
            acc = __builtin_amdgcn_wmma_f32_16x16x32_f16(
                      false, a2[t], false, bfrag, (short)0, acc, false, false);
        }
        float bias = b2[s * HID + 16 * j2 + m];
        float w3v  = W3[s * HID + 16 * j2 + m];
#pragma unroll
        for (int r = 0; r < 8; ++r)
            taup[r] += tanhf(acc[r] + bias) * w3v;
    }
    wait_async_and_sync();

    for (int j2 = 4; j2 < 8; ++j2) {                      // from wbuf[1]
        v8f acc = {};
#pragma unroll
        for (int t = 0; t < 4; ++t) {
            v16h bfrag = *(const v16h*)&wbuf[1][((j2 - 4) * 4 + t) * 512 + lane * 16];
            acc = __builtin_amdgcn_wmma_f32_16x16x32_f16(
                      false, a2[t], false, bfrag, (short)0, acc, false, false);
        }
        float bias = b2[s * HID + 16 * j2 + m];
        float w3v  = W3[s * HID + 16 * j2 + m];
#pragma unroll
        for (int r = 0; r < 8; ++r)
            taup[r] += tanhf(acc[r] + bias) * w3v;
    }

    // reduce the 128-wide dot across the 16 lanes of each half-wave
#pragma unroll
    for (int r = 0; r < 8; ++r) {
        float v = taup[r];
#pragma unroll
        for (int off = 1; off < 16; off <<= 1)
            v += __shfl_xor(v, off, 32);
        taup[r] = v;
    }

    if (m == 0) {  // lanes 0 and 16 write rows 0..7 and 8..15
        float bb = b3[s];
#pragma unroll
        for (int r = 0; r < 8; ++r)
            tau_out[s * NPTS + pbase + r + 8 * hi] = taup[r] + bb;
    }
}

// ---------------------------------------------------------------------------
// Kernel 3: cosine-window blend over the 16 tiles.
// ---------------------------------------------------------------------------
__global__ void combine(const float* __restrict__ x,
                        const float* __restrict__ mids,
                        const float* __restrict__ widths,
                        const float* __restrict__ tau,
                        float* __restrict__ out) {
    int n = blockIdx.x * blockDim.x + threadIdx.x;
    if (n >= NPTS) return;
    float xc = x[n * 6];
    float num = 0.0f, den = 0.0f;
#pragma unroll
    for (int s = 0; s < S_TILES; ++s) {
        float mid = mids[s], w = widths[s];
        float xmin = mid - 0.5f * w, xmax = mid + 0.5f * w;
        if (xc >= xmin && xc <= xmax) {
            float xs = 2.0f * (xc - xmin) / (xmax - xmin) - 1.0f;
            float wt = 0.5f * (1.0f + __cosf(xs * 3.14159265358979323846f));
            num += wt * tau[s * NPTS + n];
            den += wt;
        }
    }
    out[n] = num / den;
}

// ---------------------------------------------------------------------------
extern "C" void kernel_launch(void* const* d_in, const int* in_sizes, int n_in,
                              void* d_out, int out_size, void* d_ws, size_t ws_size,
                              hipStream_t stream) {
    const float* x      = (const float*)d_in[0];
    const float* Bmat   = (const float*)d_in[1];
    const float* W1     = (const float*)d_in[2];
    const float* b1     = (const float*)d_in[3];
    const float* W2     = (const float*)d_in[4];
    const float* b2     = (const float*)d_in[5];
    const float* W3     = (const float*)d_in[6];
    const float* b3     = (const float*)d_in[7];
    const float* mids   = (const float*)d_in[8];
    const float* widths = (const float*)d_in[9];
    float* out = (float*)d_out;

    _Float16* ws1 = (_Float16*)d_ws;
    _Float16* ws2 = ws1 + W1_FRAG_HALVES;
    float*    tau = (float*)(ws2 + W2_FRAG_HALVES);

    int prep_threads = W1_FRAG_HALVES + W2_FRAG_HALVES;
    prep_weights<<<(prep_threads + 255) / 256, 256, 0, stream>>>(W1, W2, ws1, ws2);

    int units = S_TILES * (NPTS / 16);   // 32768 wave-units, 4 waves/block
    mlp_tiles<<<units / 4, 128, 0, stream>>>(x, Bmat, ws1, b1, ws2, b2,
                                             W3, b3, mids, widths, tau);

    combine<<<(NPTS + 255) / 256, 256, 0, stream>>>(x, mids, widths, tau, out);
}
